// NER_87531433492661
// MI455X (gfx1250) — matrix operand order repaired and checked by reading, not compiled, not run on hardware
//
#include <hip/hip_runtime.h>
#include <hip/hip_bf16.h>
#include <math.h>

// Model dims (fixed by the reference)
#define Bn 16
#define Ln 512
#define Dn 512
#define NHn 8
#define HDn 64
#define Tn 17

typedef __attribute__((ext_vector_type(16))) _Float16 v16h;
typedef __attribute__((ext_vector_type(8)))  _Float16 v8h;
typedef __attribute__((ext_vector_type(8)))  float    v8f;

// ---------------------------------------------------------------------------
// GEMM:  C[M,N] = A[M,K] @ W[K,N] + bias
//   A  : f16 row-major (M x K)
//   Wt : f16 row-major TRANSPOSED weights, zero-PADDED to Npad=ceil(N/64)*64
//        rows -> all B tile loads are unconditional contiguous 32B loads.
//   C  : f32 row-major (ragged N handled on the store side only)
// One wave32 computes a 16x64 output strip: 4 independent accumulators share
// one A tile (4x A reuse, 4 independent WMMA chains per k-step).
// Control flow is wave-uniform around the WMMA (EXEC all ones).
// ---------------------------------------------------------------------------
__global__ void gemm_wmma_kernel(const _Float16* __restrict__ A,
                                 const _Float16* __restrict__ Wt,
                                 const float* __restrict__ bias,
                                 float* __restrict__ C,
                                 int M, int N, int K) {
    const int nStrips = (N + 63) >> 6;               // 64-wide output strips
    const int mTiles  = M >> 4;
    const int wid = blockIdx.x * (blockDim.x >> 5) + (threadIdx.x >> 5);
    if (wid >= mTiles * nStrips) return;

    const int tm   = wid / nStrips;
    const int tn   = wid % nStrips;
    const int lane = threadIdx.x & 31;
    const int half = lane >> 4;
    const int r    = lane & 15;

    const int row  = tm * 16 + r;                    // A row for this lane
    const int col0 = tn * 64 + r;                    // first sub-tile column

    const _Float16* __restrict__ Ar = A + (size_t)row * K;
    const _Float16* __restrict__ B0 = Wt + (size_t)(col0     ) * K;
    const _Float16* __restrict__ B1 = Wt + (size_t)(col0 + 16) * K;
    const _Float16* __restrict__ B2 = Wt + (size_t)(col0 + 32) * K;
    const _Float16* __restrict__ B3 = Wt + (size_t)(col0 + 48) * K;

    v8f acc0 = {}, acc1 = {}, acc2 = {}, acc3 = {};
    #pragma unroll 2
    for (int k0 = 0; k0 < K; k0 += 32) {
        __builtin_prefetch(Ar + k0 + 128, 0, 3);     // global_prefetch_b8

        // A tile 16x32 f16: lanes 0-15 -> K [k0,k0+16), lanes 16-31 -> K [k0+16,k0+32)
        const v16h a = *(const v16h*)(Ar + k0 + 16 * half);

        // B tiles 32x16 f16: lanes 0-15 hold K = k0..k0+15, lanes 16-31 hold +16.
        // Transposed+padded weights -> one contiguous 32-byte load each.
        const int kb = k0 + 16 * half;
        const v16h b0 = *(const v16h*)(B0 + kb);
        const v16h b1 = *(const v16h*)(B1 + kb);
        const v16h b2 = *(const v16h*)(B2 + kb);
        const v16h b3 = *(const v16h*)(B3 + kb);

        acc0 = __builtin_amdgcn_wmma_f32_16x16x32_f16(false, a, false, b0,
                                                      (short)0, acc0, false, false);
        acc1 = __builtin_amdgcn_wmma_f32_16x16x32_f16(false, a, false, b1,
                                                      (short)0, acc1, false, false);
        acc2 = __builtin_amdgcn_wmma_f32_16x16x32_f16(false, a, false, b2,
                                                      (short)0, acc2, false, false);
        acc3 = __builtin_amdgcn_wmma_f32_16x16x32_f16(false, a, false, b3,
                                                      (short)0, acc3, false, false);
    }

    #pragma unroll
    for (int j = 0; j < 4; ++j) {
        const int col = col0 + j * 16;
        if (col < N) {
            const v8f acc = (j == 0) ? acc0 : (j == 1) ? acc1 : (j == 2) ? acc2 : acc3;
            const float bb = bias ? bias[col] : 0.f;
            #pragma unroll
            for (int v = 0; v < 8; ++v) {
                const int orow = tm * 16 + v + 8 * half;   // C/D layout: M = v + 8*half
                C[(size_t)orow * N + col] = acc[v] + bb;
            }
        }
    }
}

// Wt[n*K + k] = (f16) W[k*N + n] for n < N, else 0 (padding rows)
__global__ void transpose_w_kernel(const float* __restrict__ W,
                                   _Float16* __restrict__ Wt,
                                   int K, int N, int Npad) {
    const int gid = blockIdx.x * blockDim.x + threadIdx.x;
    if (gid >= Npad * K) return;
    const int n = gid / K, k = gid - n * K;
    Wt[gid] = (n < N) ? (_Float16)W[(size_t)k * N + n] : (_Float16)0.f;
}

// e16/h16[t,d] = (f16) char_emb[tokens[t], d]
__global__ void embed_kernel(const int* __restrict__ tokens,
                             const float* __restrict__ emb,
                             _Float16* __restrict__ e16,
                             _Float16* __restrict__ h16) {
    const size_t gid = (size_t)blockIdx.x * blockDim.x + threadIdx.x;
    const size_t total = (size_t)Bn * Ln * Dn;
    if (gid >= total) return;
    const size_t t = gid >> 9;
    const int    d = (int)(gid & 511);
    const _Float16 v = (_Float16)emb[(size_t)tokens[t] * Dn + d];
    e16[gid] = v;
    h16[gid] = v;
}

// s16[b,d] = mean_l e16[b,l,d]  (f32 accumulate)
__global__ void mean_kernel(const _Float16* __restrict__ e16,
                            _Float16* __restrict__ s16) {
    const int gid = blockIdx.x * blockDim.x + threadIdx.x;   // Bn*Dn
    if (gid >= Bn * Dn) return;
    const int b = gid >> 9, d = gid & 511;
    float acc = 0.f;
    for (int l = 0; l < Ln; ++l)
        acc += (float)e16[((size_t)b * Ln + l) * Dn + d];
    s16[gid] = (_Float16)(acc * (1.f / (float)Ln));
}

// ---------------------------------------------------------------------------
// sat attention: 1 query over 5 keys per (token, head); keys/values are
// permuted rows of pre-projected Kh/Ke/Ks (Vh/Ve/Vs). 1 thread per task.
// ---------------------------------------------------------------------------
__global__ void sat_attn_kernel(const float* __restrict__ Qh,
                                const float* __restrict__ Kh,
                                const float* __restrict__ Ke,
                                const float* __restrict__ Ks,
                                const float* __restrict__ Vh,
                                const float* __restrict__ Ve,
                                const float* __restrict__ Vs,
                                _Float16* __restrict__ ctx) {
    const int gid = blockIdx.x * blockDim.x + threadIdx.x;
    if (gid >= Bn * Ln * NHn) return;
    const int head = gid & (NHn - 1);
    const int t    = gid >> 3;
    const int b    = t / Ln, l = t - b * Ln;
    const int off  = head * HDn;

    const size_t t_last = (size_t)b * Ln + ((l + 1) % Ln);
    const size_t t_next = (size_t)b * Ln + (l == 0 ? (Ln - 1) : 0);
    const size_t ts     = (size_t)t;

    const float* kp[5] = { Kh + t_last * Dn + off, Kh + ts * Dn + off,
                           Kh + t_next * Dn + off, Ke + ts * Dn + off,
                           Ks + (size_t)b * Dn + off };
    const float* vp[5] = { Vh + t_last * Dn + off, Vh + ts * Dn + off,
                           Vh + t_next * Dn + off, Ve + ts * Dn + off,
                           Vs + (size_t)b * Dn + off };

    float q[HDn];
    const float* qp = Qh + ts * Dn + off;
    #pragma unroll 8
    for (int i = 0; i < HDn; ++i) q[i] = qp[i];

    float sc[5], mx = -1e30f;
    #pragma unroll
    for (int j = 0; j < 5; ++j) {
        float d = 0.f;
        for (int i = 0; i < HDn; ++i) d += q[i] * kp[j][i];
        sc[j] = d * 0.125f;                      // 1/sqrt(64)
        mx = fmaxf(mx, sc[j]);
    }
    float sum = 0.f;
    #pragma unroll
    for (int j = 0; j < 5; ++j) { sc[j] = __expf(sc[j] - mx); sum += sc[j]; }
    const float inv = 1.f / sum;

    _Float16* op = ctx + ts * Dn + off;
    for (int i = 0; i < HDn; ++i) {
        float o = 0.f;
        #pragma unroll
        for (int j = 0; j < 5; ++j) o += sc[j] * vp[j][i];
        op[i] = (_Float16)(o * inv);
    }
}

// ---------------------------------------------------------------------------
// rel attention: per batch b, query s over 513 keys [s ; h(0..511)].
// One block per b (256 threads = 8 wave32), one wave per head.
// ---------------------------------------------------------------------------
__global__ void rel_attn_kernel(const float* __restrict__ sq,
                                const float* __restrict__ Ksr,
                                const float* __restrict__ Khr,
                                const float* __restrict__ Vsr,
                                const float* __restrict__ Vhr,
                                _Float16* __restrict__ rctx) {
    __shared__ float sc[NHn][Ln + 8];
    const int b    = blockIdx.x;
    const int head = threadIdx.x >> 5;
    const int lane = threadIdx.x & 31;
    const int off  = head * HDn;
    const int NK   = Ln + 1;                       // 513 keys

    float q[HDn];
    const float* qp = sq + (size_t)b * Dn + off;
    #pragma unroll 8
    for (int i = 0; i < HDn; ++i) q[i] = qp[i];

    for (int j = lane; j < NK; j += 32) {
        const float* k = (j == 0) ? (Ksr + (size_t)b * Dn + off)
                                  : (Khr + ((size_t)b * Ln + (j - 1)) * Dn + off);
        float d = 0.f;
        for (int i = 0; i < HDn; ++i) d += q[i] * k[i];
        sc[head][j] = d * 0.125f;
    }
    __syncthreads();

    float mx = -1e30f;
    for (int j = lane; j < NK; j += 32) mx = fmaxf(mx, sc[head][j]);
    for (int o = 16; o; o >>= 1) mx = fmaxf(mx, __shfl_xor(mx, o));

    float ssum = 0.f;
    for (int j = lane; j < NK; j += 32) {
        const float e = __expf(sc[head][j] - mx);
        sc[head][j] = e;
        ssum += e;
    }
    for (int o = 16; o; o >>= 1) ssum += __shfl_xor(ssum, o);
    const float inv = 1.f / ssum;
    __syncthreads();

    for (int dbase = 0; dbase < HDn; dbase += 32) {
        const int d = dbase + lane;
        float acc = sc[head][0] * Vsr[(size_t)b * Dn + off + d];
        for (int j = 1; j < NK; ++j)
            acc += sc[head][j] * Vhr[((size_t)b * Ln + (j - 1)) * Dn + off + d];
        rctx[(size_t)b * Dn + off + d] = (_Float16)(acc * inv);
    }
}

// ---------------------------------------------------------------------------
// y16 = (f16)( ln_w * (relu(x) - mean) * rsqrt(var + 1e-12) + ln_b )
// One block (256 threads) per 512-element row.
// ---------------------------------------------------------------------------
__global__ void relu_ln_kernel(const float* __restrict__ x,
                               const float* __restrict__ w,
                               const float* __restrict__ b,
                               _Float16* __restrict__ y) {
    __shared__ float red[256];
    const int row = blockIdx.x, tid = threadIdx.x;
    const float* xr = x + (size_t)row * Dn;

    const float v0 = fmaxf(xr[tid], 0.f);
    const float v1 = fmaxf(xr[tid + 256], 0.f);

    red[tid] = v0 + v1;
    __syncthreads();
    for (int o = 128; o; o >>= 1) { if (tid < o) red[tid] += red[tid + o]; __syncthreads(); }
    const float u = red[0] * (1.f / (float)Dn);
    __syncthreads();

    const float d0 = v0 - u, d1 = v1 - u;
    red[tid] = d0 * d0 + d1 * d1;
    __syncthreads();
    for (int o = 128; o; o >>= 1) { if (tid < o) red[tid] += red[tid + o]; __syncthreads(); }
    const float inv = rsqrtf(red[0] * (1.f / (float)Dn) + 1e-12f);

    _Float16* yr = y + (size_t)row * Dn;
    yr[tid]       = (_Float16)(w[tid]       * d0 * inv + b[tid]);
    yr[tid + 256] = (_Float16)(w[tid + 256] * d1 * inv + b[tid + 256]);
}

// ---------------------------------------------------------------------------
static inline void launch_gemm(const _Float16* A, const _Float16* Wt,
                               const float* bias, float* C,
                               int M, int N, int K, hipStream_t stream) {
    const int waves  = (M >> 4) * ((N + 63) >> 6);
    const int blocks = (waves + 7) / 8;              // 8 wave32 per 256-thread block
    gemm_wmma_kernel<<<blocks, 256, 0, stream>>>(A, Wt, bias, C, M, N, K);
}

static inline void launch_transpose(const float* W, _Float16* Wt, int K, int N,
                                    hipStream_t stream) {
    const int Npad  = ((N + 63) / 64) * 64;
    const int elems = Npad * K;
    transpose_w_kernel<<<(elems + 255) / 256, 256, 0, stream>>>(W, Wt, K, N, Npad);
}

extern "C" void kernel_launch(void* const* d_in, const int* in_sizes, int n_in,
                              void* d_out, int out_size, void* d_ws, size_t ws_size,
                              hipStream_t stream) {
    (void)in_sizes; (void)n_in; (void)out_size; (void)ws_size;

    const int*   tokens   = (const int*)  d_in[0];
    const float* char_emb = (const float*)d_in[4];
    const float* sat_qw = (const float*)d_in[5],  *sat_qb = (const float*)d_in[6];
    const float* sat_kw = (const float*)d_in[7],  *sat_kb = (const float*)d_in[8];
    const float* sat_vw = (const float*)d_in[9],  *sat_vb = (const float*)d_in[10];
    const float* sat_ow = (const float*)d_in[11], *sat_ob = (const float*)d_in[12];
    const float* rel_qw = (const float*)d_in[13], *rel_qb = (const float*)d_in[14];
    const float* rel_kw = (const float*)d_in[15], *rel_kb = (const float*)d_in[16];
    const float* rel_vw = (const float*)d_in[17], *rel_vb = (const float*)d_in[18];
    const float* rel_ow = (const float*)d_in[19], *rel_ob = (const float*)d_in[20];
    const float* ln_sat_w = (const float*)d_in[21], *ln_sat_b = (const float*)d_in[22];
    const float* ln_rel_w = (const float*)d_in[23], *ln_rel_b = (const float*)d_in[24];
    const float* ofc_w  = (const float*)d_in[25], *ofc_b  = (const float*)d_in[26];

    // ---------------- workspace layout ----------------
    const size_t BIG = (size_t)Bn * Ln * Dn;         // 4,194,304 elems
    const size_t SB  = (size_t)Bn * Dn;              // 8,192 elems
    const size_t WSZ = (size_t)Dn * Dn;              // 262,144 elems

    char* p = (char*)d_ws;
    // f32 big buffers (GEMM outputs consumed by attention / LN)
    float* Ke  = (float*)p; p += BIG * 4;
    float* Ve  = (float*)p; p += BIG * 4;
    float* Qh  = (float*)p; p += BIG * 4;
    float* Kh  = (float*)p; p += BIG * 4;
    float* Vh  = (float*)p; p += BIG * 4;
    // f16 big buffers (GEMM A-inputs)
    _Float16* e16   = (_Float16*)p; p += BIG * 2;
    _Float16* h16   = (_Float16*)p; p += BIG * 2;
    _Float16* ctx16 = (_Float16*)p; p += BIG * 2;
    // f16 transposed weights (Npad x K)
    _Float16* satq_t = (_Float16*)p; p += WSZ * 2;
    _Float16* satk_t = (_Float16*)p; p += WSZ * 2;
    _Float16* satv_t = (_Float16*)p; p += WSZ * 2;
    _Float16* sato_t = (_Float16*)p; p += WSZ * 2;
    _Float16* relq_t = (_Float16*)p; p += WSZ * 2;
    _Float16* relk_t = (_Float16*)p; p += WSZ * 2;
    _Float16* relv_t = (_Float16*)p; p += WSZ * 2;
    _Float16* relo_t = (_Float16*)p; p += WSZ * 2;
    _Float16* ofc_t  = (_Float16*)p; p += (size_t)64 * Dn * 2;   // Npad=64, 17 rows used
    // small buffers
    float* sq   = (float*)p; p += SB * 4;
    float* Ks   = (float*)p; p += SB * 4;
    float* Vs   = (float*)p; p += SB * 4;
    float* Ksr  = (float*)p; p += SB * 4;
    float* Vsr  = (float*)p; p += SB * 4;
    float* rout = (float*)p; p += SB * 4;
    _Float16* s16    = (_Float16*)p; p += SB * 2;
    _Float16* rctx16 = (_Float16*)p; p += SB * 2;

    const int MB = Bn * Ln;                          // 8192 rows

    // Pre-transpose + f16-convert all weights (resident in L2 thereafter)
    launch_transpose(sat_qw, satq_t, Dn, Dn, stream);
    launch_transpose(sat_kw, satk_t, Dn, Dn, stream);
    launch_transpose(sat_vw, satv_t, Dn, Dn, stream);
    launch_transpose(sat_ow, sato_t, Dn, Dn, stream);
    launch_transpose(rel_qw, relq_t, Dn, Dn, stream);
    launch_transpose(rel_kw, relk_t, Dn, Dn, stream);
    launch_transpose(rel_vw, relv_t, Dn, Dn, stream);
    launch_transpose(rel_ow, relo_t, Dn, Dn, stream);
    launch_transpose(ofc_w,  ofc_t,  Dn, Tn, stream);

    // Embedding + running summary
    embed_kernel<<<(int)(BIG / 256), 256, 0, stream>>>(tokens, char_emb, e16, h16);
    mean_kernel<<<(Bn * Dn) / 256, 256, 0, stream>>>(e16, s16);

    // K/V of e (constant across cycles)
    launch_gemm(e16, satk_t, sat_kb, Ke, MB, Dn, Dn, stream);
    launch_gemm(e16, satv_t, sat_vb, Ve, MB, Dn, Dn, stream);

    for (int cyc = 0; cyc < 2; ++cyc) {
        // ---- sat attention (1 query over 5 permuted keys per token) ----
        launch_gemm(h16, satq_t, sat_qb, Qh, MB, Dn, Dn, stream);
        launch_gemm(h16, satk_t, sat_kb, Kh, MB, Dn, Dn, stream);
        launch_gemm(h16, satv_t, sat_vb, Vh, MB, Dn, Dn, stream);
        launch_gemm(s16, satk_t, sat_kb, Ks, Bn, Dn, Dn, stream);
        launch_gemm(s16, satv_t, sat_vb, Vs, Bn, Dn, Dn, stream);
        sat_attn_kernel<<<(MB * NHn) / 256, 256, 0, stream>>>(Qh, Kh, Ke, Ks,
                                                              Vh, Ve, Vs, ctx16);
        launch_gemm(ctx16, sato_t, sat_ob, Qh, MB, Dn, Dn, stream);   // reuse Qh
        relu_ln_kernel<<<MB, 256, 0, stream>>>(Qh, ln_sat_w, ln_sat_b, h16);

        // ---- rel attention (16 queries over [s ; h] = 513 keys) ----
        launch_gemm(s16, relq_t, rel_qb, sq,  Bn, Dn, Dn, stream);
        launch_gemm(h16, relk_t, rel_kb, Kh,  MB, Dn, Dn, stream);    // reuse Kh/Vh
        launch_gemm(h16, relv_t, rel_vb, Vh,  MB, Dn, Dn, stream);
        launch_gemm(s16, relk_t, rel_kb, Ksr, Bn, Dn, Dn, stream);
        launch_gemm(s16, relv_t, rel_vb, Vsr, Bn, Dn, Dn, stream);
        rel_attn_kernel<<<Bn, 256, 0, stream>>>(sq, Ksr, Kh, Vsr, Vh, rctx16);
        launch_gemm(rctx16, relo_t, rel_ob, rout, Bn, Dn, Dn, stream);
        relu_ln_kernel<<<Bn, 256, 0, stream>>>(rout, ln_rel_w, ln_rel_b, s16);
    }

    // logits = h @ ofc_w + ofc_b   (ragged N = 17)
    launch_gemm(h16, ofc_t, ofc_b, (float*)d_out, MB, Tn, Dn, stream);
}
